// DirGATv2Conv_32212254720660
// MI455X (gfx1250) — matrix operand order repaired and checked by reading, not compiled, not run on hardware
//
#include <hip/hip_runtime.h>
#include <hip/hip_bf16.h>

typedef __attribute__((ext_vector_type(16))) _Float16 v16h;
typedef __attribute__((ext_vector_type(4)))  _Float16 v4h;
typedef __attribute__((ext_vector_type(8)))  float    v8f;

#define N_  50000
#define E_  800000
#define D_  128
#define H_  4
#define C_  32
#define HC_ 128
#define ED_ 16
#define NEG_SLOPE_ 0.2f
#define ALPHA_ 0.5f
#define MT_ (N_ / 16)          // 3125 node row-tiles
#define EDGE_BLOCKS_ 8192      // 65536 waves, grid-stride over edges

// ---------- float <-> orderable-uint key (for atomic max on f32) ----------
__device__ __forceinline__ unsigned fkey(float f) {
    unsigned u = __float_as_uint(f);
    return (u & 0x80000000u) ? ~u : (u | 0x80000000u);
}
__device__ __forceinline__ float fdecode(unsigned k) {
    unsigned u = (k & 0x80000000u) ? (k & 0x7fffffffu) : ~k;
    return __uint_as_float(u);
}

// ---------------------- zero-fill scratch ----------------------
__global__ __launch_bounds__(256) void zero_kernel(unsigned* __restrict__ p, size_t n) {
    size_t i = (size_t)blockIdx.x * 256 + threadIdx.x;
    if (i < n) p[i] = 0u;
}

// -------- transpose + f32->f16 convert the 4 node weight matrices --------
// Wt[widx][n][k] = W[widx][k][n]
__global__ __launch_bounds__(256) void prep_weights_kernel(
    const float* __restrict__ Wl1, const float* __restrict__ Wr1,
    const float* __restrict__ Wl2, const float* __restrict__ Wr2,
    _Float16* __restrict__ Wt) {
    int idx = blockIdx.x * 256 + threadIdx.x;       // 4*128*128 = 65536
    int widx = idx >> 14;
    int rem  = idx & 16383;
    int n = rem >> 7;
    int k = rem & 127;
    const float* W = (widx == 0) ? Wl1 : (widx == 1) ? Wr1 : (widx == 2) ? Wl2 : Wr2;
    Wt[idx] = (_Float16)W[k * 128 + n];
}

// -------------- node linear layers via WMMA f16 (f32 accumulate) --------------
// one wave per 16-row node tile; computes xl1/xr1/xl2/xr2 (f16 outputs)
__global__ __launch_bounds__(256) void node_gemm_kernel(
    const float* __restrict__ x,
    const _Float16* __restrict__ Wt,     // [4][128][128] transposed f16
    const float* __restrict__ bl1, const float* __restrict__ br1,
    const float* __restrict__ bl2, const float* __restrict__ br2,
    _Float16* __restrict__ xl1, _Float16* __restrict__ xr1,
    _Float16* __restrict__ xl2, _Float16* __restrict__ xr2) {
    const int wid  = threadIdx.x >> 5;
    const int lane = threadIdx.x & 31;
    const int mt = blockIdx.x * 8 + wid;
    if (mt >= MT_) return;                    // wave-uniform; EXEC stays full for WMMA
    const int m0 = mt * 16;
    const int lx = lane & 15;
    const int h  = lane >> 4;

    // A fragments: x tile 16x128 split into 4 K-slices of 32 (lane half h holds 16 contiguous K: 16h..16h+15)
    const float* xrow = x + (size_t)(m0 + lx) * D_;
    v16h afr[4];
#pragma unroll
    for (int kk = 0; kk < 4; ++kk) {
        const float* base = xrow + kk * 32 + 16 * h;
        float4 a0 = *(const float4*)(base);
        float4 a1 = *(const float4*)(base + 4);
        float4 b0 = *(const float4*)(base + 8);
        float4 b1 = *(const float4*)(base + 12);
        v16h v;
        v[0]=(_Float16)a0.x; v[1]=(_Float16)a0.y; v[2]=(_Float16)a0.z; v[3]=(_Float16)a0.w;
        v[4]=(_Float16)a1.x; v[5]=(_Float16)a1.y; v[6]=(_Float16)a1.z; v[7]=(_Float16)a1.w;
        v[8]=(_Float16)b0.x; v[9]=(_Float16)b0.y; v[10]=(_Float16)b0.z; v[11]=(_Float16)b0.w;
        v[12]=(_Float16)b1.x; v[13]=(_Float16)b1.y; v[14]=(_Float16)b1.z; v[15]=(_Float16)b1.w;
        afr[kk] = v;
    }

    const float* biases[4] = {bl1, br1, bl2, br2};
    _Float16*    outs[4]   = {xl1, xr1, xl2, xr2};

#pragma unroll
    for (int widx = 0; widx < 4; ++widx) {
        const _Float16* wbase = Wt + (size_t)widx * (128 * 128);
        const float* bvec = biases[widx];
        _Float16* op = outs[widx];
        for (int nt = 0; nt < 8; ++nt) {
            const int n0 = nt * 16;
            v8f acc = {};
#pragma unroll
            for (int kk = 0; kk < 4; ++kk) {
                // B fragment: lane holds col n0+lx, 16 contiguous K halves (32B aligned)
                const _Float16* wp = wbase + (size_t)(n0 + lx) * 128 + kk * 32 + 16 * h;
                v16h bfr = *(const v16h*)wp;
                acc = __builtin_amdgcn_wmma_f32_16x16x32_f16(
                    false, afr[kk], false, bfr, (short)0, acc, false, false);
            }
            const float bn = bvec[n0 + lx];
#pragma unroll
            for (int r = 0; r < 8; ++r) {
                const int row = m0 + r + 8 * h;      // D layout: vgpr r, lanes>=16 -> M+8
                op[(size_t)row * HC_ + n0 + lx] = (_Float16)(acc[r] + bn);
            }
        }
    }
}

// -------------- pass 1: per-edge attention scores + segment max --------------
// one wave per edge per loop iteration; lane covers 4 channels
__global__ __launch_bounds__(256) void edge_score_kernel(
    const long long* __restrict__ sIdx, const long long* __restrict__ tIdx,
    const _Float16* __restrict__ xl, const _Float16* __restrict__ xr,
    const float* __restrict__ edge_attr, const float* __restrict__ We,
    const float* __restrict__ att,
    float* __restrict__ scoreOut, unsigned* __restrict__ smaxKeys) {
    __shared__ float sWe[ED_ * HC_];   // 8 KB
    __shared__ float sAtt[HC_];
    for (int i = threadIdx.x; i < ED_ * HC_; i += 256) sWe[i] = We[i];
    if (threadIdx.x < HC_) sAtt[threadIdx.x] = att[threadIdx.x];
    __syncthreads();

    const int lane = threadIdx.x & 31;
    const int estart = blockIdx.x * 8 + (threadIdx.x >> 5);
    const int estep  = EDGE_BLOCKS_ * 8;

    for (int edge = estart; edge < E_; edge += estep) {
        const long long s = sIdx[edge];
        const long long t = tIdx[edge];

        v4h xlv = *(const v4h*)(xl + (size_t)s * HC_ + lane * 4);
        v4h xrv = *(const v4h*)(xr + (size_t)t * HC_ + lane * 4);

        // edge_attr broadcast via shuffle, e = ea @ We from LDS
        float eav = (lane < ED_) ? edge_attr[(size_t)edge * ED_ + lane] : 0.f;
        float ec0 = 0.f, ec1 = 0.f, ec2 = 0.f, ec3 = 0.f;
#pragma unroll
        for (int d = 0; d < ED_; ++d) {
            float ad = __shfl(eav, d, 32);
            const float* w = &sWe[d * HC_ + lane * 4];
            ec0 += ad * w[0]; ec1 += ad * w[1]; ec2 += ad * w[2]; ec3 += ad * w[3];
        }

        float ec[4] = {ec0, ec1, ec2, ec3};
        float partial = 0.f;
#pragma unroll
        for (int j = 0; j < 4; ++j) {
            float m = (float)xlv[j] + (float)xrv[j] + ec[j];
            m = (m > 0.f) ? m : NEG_SLOPE_ * m;
            partial += m * sAtt[lane * 4 + j];
        }
        // reduce 8 lanes (one head = 32 channels = 8 lanes)
        partial += __shfl_xor(partial, 1, 32);
        partial += __shfl_xor(partial, 2, 32);
        partial += __shfl_xor(partial, 4, 32);

        if ((lane & 7) == 0) {
            const int h = lane >> 3;
            scoreOut[(size_t)edge * H_ + h] = partial;
            atomicMax(&smaxKeys[(size_t)t * H_ + h], fkey(partial));
        }
    }
}

// -------- pass 2: exp, denominator, weighted scatter-add of numerator --------
__global__ __launch_bounds__(256) void edge_aggr_kernel(
    const long long* __restrict__ sIdx, const long long* __restrict__ tIdx,
    const _Float16* __restrict__ xl,
    const float* __restrict__ scoreIn, const unsigned* __restrict__ smaxKeys,
    float* __restrict__ den, float* __restrict__ num) {
    const int lane = threadIdx.x & 31;
    const int h = lane >> 3;
    const int estart = blockIdx.x * 8 + (threadIdx.x >> 5);
    const int estep  = EDGE_BLOCKS_ * 8;

    for (int edge = estart; edge < E_; edge += estep) {
        const long long s = sIdx[edge];
        const long long t = tIdx[edge];

        const float score = scoreIn[(size_t)edge * H_ + h];
        const float smax  = fdecode(smaxKeys[(size_t)t * H_ + h]);
        const float a = __expf(score - smax);

        if ((lane & 7) == 0) atomicAdd(&den[(size_t)t * H_ + h], a);

        v4h xlv = *(const v4h*)(xl + (size_t)s * HC_ + lane * 4);
        float* np = num + (size_t)t * HC_ + lane * 4;
        atomicAdd(np + 0, a * (float)xlv[0]);
        atomicAdd(np + 1, a * (float)xlv[1]);
        atomicAdd(np + 2, a * (float)xlv[2]);
        atomicAdd(np + 3, a * (float)xlv[3]);
    }
}

// -------------- per-node finalize: divide, bias, alpha blend --------------
__global__ __launch_bounds__(256) void finalize_kernel(
    const float* __restrict__ num1, const float* __restrict__ den1, const float* __restrict__ bias1,
    const float* __restrict__ num2, const float* __restrict__ den2, const float* __restrict__ bias2,
    float* __restrict__ out) {
    size_t idx = (size_t)blockIdx.x * 256 + threadIdx.x;
    if (idx >= (size_t)N_ * HC_) return;
    const int hc = (int)(idx & (HC_ - 1));
    const size_t n = idx >> 7;
    const int h = hc >> 5;
    const float o1 = num1[idx] / (den1[n * H_ + h] + 1e-16f) + bias1[hc];
    const float o2 = num2[idx] / (den2[n * H_ + h] + 1e-16f) + bias2[hc];
    out[idx] = (1.0f - ALPHA_) * o1 + ALPHA_ * o2;
}

extern "C" void kernel_launch(void* const* d_in, const int* in_sizes, int n_in,
                              void* d_out, int out_size, void* d_ws, size_t ws_size,
                              hipStream_t stream) {
    const float*     x   = (const float*)d_in[0];
    const long long* ei  = (const long long*)d_in[1];   // int64 edge_index [2][E]
    const float*     ea  = (const float*)d_in[2];
    const float *Wl1 = (const float*)d_in[3],  *bl1 = (const float*)d_in[4];
    const float *Wr1 = (const float*)d_in[5],  *br1 = (const float*)d_in[6];
    const float *We1 = (const float*)d_in[7],  *att1 = (const float*)d_in[8];
    const float *bias1 = (const float*)d_in[9];
    const float *Wl2 = (const float*)d_in[10], *bl2 = (const float*)d_in[11];
    const float *Wr2 = (const float*)d_in[12], *br2 = (const float*)d_in[13];
    const float *We2 = (const float*)d_in[14], *att2 = (const float*)d_in[15];
    const float *bias2 = (const float*)d_in[16];
    float* out = (float*)d_out;
    char* ws = (char*)d_ws;

    // ---- workspace layout ----
    const size_t szWt   = 4ull * 128 * 128 * sizeof(_Float16);   // 131072
    const size_t szNode = (size_t)N_ * HC_ * sizeof(_Float16);   // 12.8 MB
    const size_t szScr  = (size_t)E_ * H_ * sizeof(float);       // 12.8 MB
    const size_t szSmax = (size_t)N_ * H_ * sizeof(unsigned);    // 0.8 MB
    const size_t szDen  = (size_t)N_ * H_ * sizeof(float);
    const size_t szNum  = (size_t)N_ * HC_ * sizeof(float);      // 25.6 MB

    _Float16* Wt  = (_Float16*)(ws);
    _Float16* xl1 = (_Float16*)(ws + szWt);
    _Float16* xr1 = (_Float16*)(ws + szWt + szNode);
    _Float16* xl2 = (_Float16*)(ws + szWt + 2 * szNode);
    _Float16* xr2 = (_Float16*)(ws + szWt + 3 * szNode);
    float* score1 = (float*)(ws + szWt + 4 * szNode);
    float* score2 = (float*)(ws + szWt + 4 * szNode + szScr);
    char* zbase   =          ws + szWt + 4 * szNode + 2 * szScr;
    unsigned* smax1 = (unsigned*)(zbase);
    unsigned* smax2 = (unsigned*)(zbase + szSmax);
    float* den1 = (float*)(zbase + 2 * szSmax);
    float* den2 = (float*)(zbase + 2 * szSmax + szDen);
    float* num1 = (float*)(zbase + 2 * szSmax + 2 * szDen);
    float* num2 = (float*)(zbase + 2 * szSmax + 2 * szDen + szNum);
    const size_t zeroWords = (2 * szSmax + 2 * szDen + 2 * szNum) / 4;

    const long long* srcI = ei;        // edge_index[0]
    const long long* dstI = ei + E_;   // edge_index[1]

    // 1) clear accumulators (smax key 0 == ordered -inf sentinel)
    zero_kernel<<<(unsigned)((zeroWords + 255) / 256), 256, 0, stream>>>(
        (unsigned*)zbase, zeroWords);

    // 2) weight prep (f16 transpose)
    prep_weights_kernel<<<(4 * 128 * 128) / 256, 256, 0, stream>>>(Wl1, Wr1, Wl2, Wr2, Wt);

    // 3) node projections via WMMA
    node_gemm_kernel<<<(MT_ + 7) / 8, 256, 0, stream>>>(
        x, Wt, bl1, br1, bl2, br2, xl1, xr1, xl2, xr2);

    // 4) scores + segment max (conv1: src->dst; conv2: swapped edges)
    edge_score_kernel<<<EDGE_BLOCKS_, 256, 0, stream>>>(
        srcI, dstI, xl1, xr1, ea, We1, att1, score1, smax1);
    edge_score_kernel<<<EDGE_BLOCKS_, 256, 0, stream>>>(
        dstI, srcI, xl2, xr2, ea, We2, att2, score2, smax2);

    // 5) softmax numer/denom scatter
    edge_aggr_kernel<<<EDGE_BLOCKS_, 256, 0, stream>>>(
        srcI, dstI, xl1, score1, smax1, den1, num1);
    edge_aggr_kernel<<<EDGE_BLOCKS_, 256, 0, stream>>>(
        dstI, srcI, xl2, score2, smax2, den2, num2);

    // 6) finalize + alpha blend
    finalize_kernel<<<(unsigned)(((size_t)N_ * HC_ + 255) / 256), 256, 0, stream>>>(
        num1, den1, bias1, num2, den2, bias2, out);
}
